// CausalSelfAttention_45079976739499
// MI455X (gfx1250) — matrix operand order repaired and checked by reading, not compiled, not run on hardware
//
#include <hip/hip_runtime.h>
#include <cstddef>
#include <cstdint>

// ---------------------------------------------------------------------------
// CausalSelfAttention forward for MI455X (gfx1250, wave32).
// B=4, T=2048, C=1024, H=16, HD=64.
//
// All matmuls run on v_wmma_f32_16x16x32_bf16 with split-bf16 (hi+lo)
// 3-product emulation of f32 (Ah*Bh + Ah*Bl + Al*Bh, ~f32 accuracy).
// Operands are pre-split into bf16 hi/lo planes in global memory, so the
// GEMM / attention inner loops contain no conversion VALU: fragments are
// two 16B LDS loads each. K/V and GEMM tiles are staged into LDS with
// double-buffered GLOBAL_LOAD_ASYNC_TO_LDS_B128 (ASYNCcnt).
// ---------------------------------------------------------------------------

typedef __attribute__((ext_vector_type(16))) __bf16        v16bf;
typedef __attribute__((ext_vector_type(8)))  float         v8f;
typedef __attribute__((ext_vector_type(4)))  unsigned int  u32x4;
typedef unsigned int u32;

#define BQ 4
#define TT 2048
#define CC 1024
#define NH 16
#define HD 64
#define N3 (3 * CC)
#define MM (BQ * TT)                 // 8192
#define PLANE_X   ((size_t)MM * CC)          // 8,388,608
#define PLANE_WA  ((size_t)N3 * CC)          // 3,145,728
#define PLANE_WP  ((size_t)CC * CC)          // 1,048,576
#define PLANE_HED ((size_t)BQ * NH * TT * HD)// 8,388,608

// ----- WMMA helpers --------------------------------------------------------
union FragU { v16bf v; u32x4 q[2]; };

__device__ __forceinline__ v8f wmma3(v16bf ah, v16bf al, v16bf bh, v16bf bl, v8f acc) {
    acc = __builtin_amdgcn_wmma_f32_16x16x32_bf16(false, ah, false, bh, (short)0, acc, false, false);
    acc = __builtin_amdgcn_wmma_f32_16x16x32_bf16(false, ah, false, bl, (short)0, acc, false, false);
    acc = __builtin_amdgcn_wmma_f32_16x16x32_bf16(false, al, false, bh, (short)0, acc, false, false);
    return acc;
}

__device__ __forceinline__ void split1(float x, __bf16& h, __bf16& l) {
    h = (__bf16)x;
    l = (__bf16)(x - (float)h);
}

// ----- gfx1250 async global->LDS copy (ASYNCcnt) ---------------------------
// LDS offset = low 32 bits of the flat address of a __shared__ object
// (aperture rule).
__device__ __forceinline__ u32 lds_off(const void* p) {
    return (u32)(uintptr_t)p;
}
// GVS mode: mem = SGPR64 base + signed VGPR32 offset; VDST = per-lane LDS addr.
__device__ __forceinline__ void async_b128(u32 lds_addr, const void* sbase, u32 goff) {
    asm volatile("global_load_async_to_lds_b128 %0, %1, %2 offset:0"
                 :: "v"(lds_addr), "v"(goff), "s"(sbase)
                 : "memory");
}
__device__ __forceinline__ void wait_async0() {
    asm volatile("s_wait_asynccnt 0" ::: "memory");
}

// ---------------------------------------------------------------------------
// Prologue converters: f32 -> (hi,lo) bf16 planes.
// ---------------------------------------------------------------------------
__global__ __launch_bounds__(256) void split_plain(const float* __restrict__ in,
                                                   __bf16* __restrict__ oh,
                                                   __bf16* __restrict__ ol, int n4) {
    int i = blockIdx.x * 256 + threadIdx.x;
    if (i >= n4) return;
    float4 v = ((const float4*)in)[i];
    __bf16 h, l;
    split1(v.x, h, l); oh[4*i+0] = h; ol[4*i+0] = l;
    split1(v.y, h, l); oh[4*i+1] = h; ol[4*i+1] = l;
    split1(v.z, h, l); oh[4*i+2] = h; ol[4*i+2] = l;
    split1(v.w, h, l); oh[4*i+3] = h; ol[4*i+3] = l;
}

// out[n*K + k] = split(in[k*N + n])  (transpose; K = 1024 here)
__global__ __launch_bounds__(256) void split_transpose(const float* __restrict__ in,
                                                       __bf16* __restrict__ oh,
                                                       __bf16* __restrict__ ol, int N) {
    int tid = blockIdx.x * 256 + threadIdx.x;
    int k = tid & (CC - 1);
    int n = tid >> 10;
    if (n >= N) return;
    __bf16 h, l;
    split1(in[(size_t)k * N + n], h, l);
    oh[(size_t)n * CC + k] = h;
    ol[(size_t)n * CC + k] = l;
}

// ---------------------------------------------------------------------------
// Shared GEMM: D[64x64 per block] = A[M x 1024] * B^T[N x 1024]^T.
// A,B given as hi/lo bf16 planes, both with row pitch 1024 bf16 (2048 B).
// Block: 128 threads (4 waves); wave w owns rows [m0+16w, +16).
// mode 0: scatter into q/k/v split planes.  mode 1: out = D + bias (f32).
// LDS tiles: A planes [64 rows][32 k], B planes n-major [64 n][32 k],
// stride 40 bf16 (80 B): fragment = two 16B reads, bank-spread.
// ---------------------------------------------------------------------------
__global__ __launch_bounds__(128) void gemm_bf(
    const __bf16* __restrict__ Ahp, const __bf16* __restrict__ Alp,
    const __bf16* __restrict__ Bhp, const __bf16* __restrict__ Blp,
    int mode,
    __bf16* __restrict__ q_h, __bf16* __restrict__ q_l,
    __bf16* __restrict__ k_h, __bf16* __restrict__ k_l,
    __bf16* __restrict__ v_h, __bf16* __restrict__ v_l,
    const float* __restrict__ bias, float* __restrict__ outp) {

    __shared__ __attribute__((aligned(16))) __bf16 Ahs[2][64 * 40];
    __shared__ __attribute__((aligned(16))) __bf16 Als[2][64 * 40];
    __shared__ __attribute__((aligned(16))) __bf16 Bhs[2][64 * 40];
    __shared__ __attribute__((aligned(16))) __bf16 Bls[2][64 * 40];

    const int tid  = threadIdx.x;
    const int lane = tid & 31;
    const int wv   = tid >> 5;
    const int half = lane >> 4;
    const int l16  = lane & 15;
    const int m0 = blockIdx.x * 64;
    const int n0 = blockIdx.y * 64;

    v8f acc[4] = {};

    // Stage one 64x32 tile-set (4 planes) into buffer `buf` via async copies.
    auto stage = [&](int kb, int buf) {
#pragma unroll
        for (int i = 0; i < 2; ++i) {
            int c = tid + i * 128;                  // 256 chunks of 16B / plane
            int row = c >> 2, col = (c & 3) << 4;   // 64 rows x 64B
            u32 lof = (u32)(row * 80 + col);
            u32 ga  = (u32)(m0 + row) * 2048u + (u32)kb * 2u + (u32)col;
            u32 gb  = (u32)(n0 + row) * 2048u + (u32)kb * 2u + (u32)col;
            async_b128(lds_off(&Ahs[buf][0]) + lof, Ahp, ga);
            async_b128(lds_off(&Als[buf][0]) + lof, Alp, ga);
            async_b128(lds_off(&Bhs[buf][0]) + lof, Bhp, gb);
            async_b128(lds_off(&Bls[buf][0]) + lof, Blp, gb);
        }
    };

    stage(0, 0);
    for (int it = 0; it < CC / 32; ++it) {
        const int buf = it & 1;
        wait_async0();
        __syncthreads();
        if (it + 1 < CC / 32) stage((it + 1) * 32, buf ^ 1);

        // A fragments: row l16 of this wave's strip; K groups 0-7/16-23 (+8*half).
        const __bf16* abh = &Ahs[buf][(wv * 16 + l16) * 40];
        const __bf16* abl = &Als[buf][(wv * 16 + l16) * 40];
        FragU fah, fal;
        fah.q[0] = *(const u32x4*)(abh + 8 * half);
        fah.q[1] = *(const u32x4*)(abh + 16 + 8 * half);
        fal.q[0] = *(const u32x4*)(abl + 8 * half);
        fal.q[1] = *(const u32x4*)(abl + 16 + 8 * half);

#pragma unroll
        for (int nt = 0; nt < 4; ++nt) {
            const __bf16* bbh = &Bhs[buf][(nt * 16 + l16) * 40 + 8 * half];
            const __bf16* bbl = &Bls[buf][(nt * 16 + l16) * 40 + 8 * half];
            FragU fbh, fbl;
            fbh.q[0] = *(const u32x4*)(bbh);
            fbh.q[1] = *(const u32x4*)(bbh + 16);
            fbl.q[0] = *(const u32x4*)(bbl);
            fbl.q[1] = *(const u32x4*)(bbl + 16);
            acc[nt] = wmma3(fah.v, fal.v, fbh.v, fbl.v, acc[nt]);
        }
    }

    if (mode == 0) {
        // Scatter into q/k [B,H,T,HD] planes and v [B,H,HD,T] (d-major) planes.
#pragma unroll
        for (int nt = 0; nt < 4; ++nt) {
#pragma unroll
            for (int r = 0; r < 8; ++r) {
                int m = m0 + wv * 16 + r + half * 8;
                int n = n0 + nt * 16 + l16;
                int b = m >> 11, t = m & (TT - 1);
                int which = n >> 10, c = n & (CC - 1);
                int h = c >> 6, d = c & (HD - 1);
                __bf16 hh, ll;
                split1(acc[nt][r], hh, ll);
                if (which == 0) {
                    size_t idx = ((size_t)(b * NH + h) * TT + t) * HD + d;
                    q_h[idx] = hh; q_l[idx] = ll;
                } else if (which == 1) {
                    size_t idx = ((size_t)(b * NH + h) * TT + t) * HD + d;
                    k_h[idx] = hh; k_l[idx] = ll;
                } else {
                    size_t idx = ((size_t)(b * NH + h) * HD + d) * TT + t;
                    v_h[idx] = hh; v_l[idx] = ll;
                }
            }
        }
    } else {
#pragma unroll
        for (int nt = 0; nt < 4; ++nt) {
#pragma unroll
            for (int r = 0; r < 8; ++r) {
                int m = m0 + wv * 16 + r + half * 8;
                int n = n0 + nt * 16 + l16;
                outp[(size_t)m * CC + n] = acc[nt][r] + bias[n];
            }
        }
    }
}

// ---------------------------------------------------------------------------
// Flash attention per (b,h). Block: 128 threads = 4 waves, 64 query rows
// (16 per wave). 64-key blocks double-buffered in LDS via async copies.
// K planes [key][d] (natural), V planes [d][key] (pre-transposed in ws),
// so both S and PV B-fragments are two contiguous 16B runs along k.
// ---------------------------------------------------------------------------
__global__ __launch_bounds__(128) void attn_bf(
    const __bf16* __restrict__ Qhp, const __bf16* __restrict__ Qlp,
    const __bf16* __restrict__ Khp, const __bf16* __restrict__ Klp,
    const __bf16* __restrict__ Vhp, const __bf16* __restrict__ Vlp,
    __bf16* __restrict__ ctx_h, __bf16* __restrict__ ctx_l) {

    __shared__ __attribute__((aligned(16))) __bf16 Kth[2][64 * 72];
    __shared__ __attribute__((aligned(16))) __bf16 Ktl[2][64 * 72];
    __shared__ __attribute__((aligned(16))) __bf16 Vth[2][64 * 72];
    __shared__ __attribute__((aligned(16))) __bf16 Vtl[2][64 * 72];
    __shared__ __attribute__((aligned(16))) __bf16 Psh[4][16 * 72];
    __shared__ __attribute__((aligned(16))) __bf16 Psl[4][16 * 72];

    const int tid  = threadIdx.x;
    const int lane = tid & 31;
    const int wv   = tid >> 5;
    const int half = lane >> 4;
    const int l16  = lane & 15;
    const int bh = blockIdx.x;          // b*H + h
    const int q0 = blockIdx.y * 64;

    const u32 head_byte = (u32)bh * (u32)(TT * HD * 2);   // same for K and V planes

    // Q fragments: 16 rows x 64 d, split planes, no conversion.
    FragU qfh[2], qfl[2];
    {
        const __bf16* qrh = Qhp + ((size_t)bh * TT + q0 + wv * 16 + l16) * HD;
        const __bf16* qrl = Qlp + ((size_t)bh * TT + q0 + wv * 16 + l16) * HD;
#pragma unroll
        for (int kc = 0; kc < 2; ++kc) {
            qfh[kc].q[0] = *(const u32x4*)(qrh + kc * 32 + 8 * half);
            qfh[kc].q[1] = *(const u32x4*)(qrh + kc * 32 + 16 + 8 * half);
            qfl[kc].q[0] = *(const u32x4*)(qrl + kc * 32 + 8 * half);
            qfl[kc].q[1] = *(const u32x4*)(qrl + kc * 32 + 16 + 8 * half);
        }
    }

    v8f acc_o[4] = {};
    float row_m[8], row_l[8];
#pragma unroll
    for (int r = 0; r < 8; ++r) { row_m[r] = -1e30f; row_l[r] = 0.f; }

    auto stage = [&](int kb, int buf) {
#pragma unroll
        for (int i = 0; i < 4; ++i) {
            int c = tid + i * 128;                 // 512 chunks of 16B / plane
            int row = c >> 3, col = (c & 7) << 4;  // 64 rows x 128B
            u32 lof = (u32)(row * 144 + col);
            u32 gk  = head_byte + (u32)(kb + row) * 128u + (u32)col;  // K: [key][d]
            u32 gv  = head_byte + (u32)row * 4096u + (u32)kb * 2u + (u32)col; // V: [d][key]
            async_b128(lds_off(&Kth[buf][0]) + lof, Khp, gk);
            async_b128(lds_off(&Ktl[buf][0]) + lof, Klp, gk);
            async_b128(lds_off(&Vth[buf][0]) + lof, Vhp, gv);
            async_b128(lds_off(&Vtl[buf][0]) + lof, Vlp, gv);
        }
    };

    const int niter = blockIdx.y + 1;   // kb = 0,64,...,q0
    stage(0, 0);
    for (int it = 0; it < niter; ++it) {
        const int buf = it & 1;
        const int kb = it * 64;
        wait_async0();
        __syncthreads();
        if (it + 1 < niter) stage(kb + 64, buf ^ 1);

        // ---- S = Q K^T (16 x 64) ----
        v8f s[4];
#pragma unroll
        for (int nt = 0; nt < 4; ++nt) {
            v8f acc_s = {};
#pragma unroll
            for (int kc = 0; kc < 2; ++kc) {
                const __bf16* kbh = &Kth[buf][(nt * 16 + l16) * 72 + kc * 32 + 8 * half];
                const __bf16* kbl = &Ktl[buf][(nt * 16 + l16) * 72 + kc * 32 + 8 * half];
                FragU fbh, fbl;
                fbh.q[0] = *(const u32x4*)(kbh);
                fbh.q[1] = *(const u32x4*)(kbh + 16);
                fbl.q[0] = *(const u32x4*)(kbl);
                fbl.q[1] = *(const u32x4*)(kbl + 16);
                acc_s = wmma3(qfh[kc].v, qfl[kc].v, fbh.v, fbl.v, acc_s);
            }
            s[nt] = acc_s;
        }

        // ---- scale, causal mask, half-wave row-max ----
        float newm[8];
#pragma unroll
        for (int r = 0; r < 8; ++r) {
            int qrow = q0 + wv * 16 + r + half * 8;
            float mx = row_m[r];
#pragma unroll
            for (int nt = 0; nt < 4; ++nt) {
                int krow = kb + nt * 16 + l16;
                float v = s[nt][r] * 0.125f;        // 1/sqrt(64)
                v = (krow <= qrow) ? v : -1e30f;
                s[nt][r] = v;
                mx = fmaxf(mx, v);
            }
#pragma unroll
            for (int off = 1; off < 16; off <<= 1)
                mx = fmaxf(mx, __shfl_xor(mx, off, 32));
            newm[r] = mx;
        }

        // ---- exp, online stats, P -> split-bf16 LDS, rescale O ----
#pragma unroll
        for (int r = 0; r < 8; ++r) {
            float scale = __expf(row_m[r] - newm[r]);
            float partial = 0.f;
#pragma unroll
            for (int nt = 0; nt < 4; ++nt) {
                float p = __expf(s[nt][r] - newm[r]);
                partial += p;
                __bf16 ph, pl;
                split1(p, ph, pl);
                int poff = (r + half * 8) * 72 + nt * 16 + l16;
                Psh[wv][poff] = ph;
                Psl[wv][poff] = pl;
            }
#pragma unroll
            for (int off = 1; off < 16; off <<= 1)
                partial += __shfl_xor(partial, off, 32);
            row_l[r] = row_l[r] * scale + partial;
            row_m[r] = newm[r];
#pragma unroll
            for (int dt = 0; dt < 4; ++dt) acc_o[dt][r] *= scale;
        }

        // ---- O += P V (keys as K-dim, 2 chunks of 32) ----
#pragma unroll
        for (int kc = 0; kc < 2; ++kc) {
            const __bf16* pah = &Psh[wv][l16 * 72 + kc * 32 + 8 * half];
            const __bf16* pal = &Psl[wv][l16 * 72 + kc * 32 + 8 * half];
            FragU fph, fpl;
            fph.q[0] = *(const u32x4*)(pah);
            fph.q[1] = *(const u32x4*)(pah + 16);
            fpl.q[0] = *(const u32x4*)(pal);
            fpl.q[1] = *(const u32x4*)(pal + 16);
#pragma unroll
            for (int dt = 0; dt < 4; ++dt) {
                const __bf16* vbh = &Vth[buf][(dt * 16 + l16) * 72 + kc * 32 + 8 * half];
                const __bf16* vbl = &Vtl[buf][(dt * 16 + l16) * 72 + kc * 32 + 8 * half];
                FragU fvh, fvl;
                fvh.q[0] = *(const u32x4*)(vbh);
                fvh.q[1] = *(const u32x4*)(vbh + 16);
                fvl.q[0] = *(const u32x4*)(vbl);
                fvl.q[1] = *(const u32x4*)(vbl + 16);
                acc_o[dt] = wmma3(fph.v, fpl.v, fvh.v, fvl.v, acc_o[dt]);
            }
        }
        __syncthreads();
    }

    // Epilogue: O /= l, write ctx split planes in [B*T, C] layout (c = h*64+d).
    const int b = bh >> 4, h = bh & 15;
#pragma unroll
    for (int r = 0; r < 8; ++r) {
        float inv = 1.f / row_l[r];
        int t = q0 + wv * 16 + r + half * 8;
#pragma unroll
        for (int dt = 0; dt < 4; ++dt) {
            int d = dt * 16 + l16;
            size_t idx = (size_t)(b * TT + t) * CC + h * HD + d;
            __bf16 hh, ll;
            split1(acc_o[dt][r] * inv, hh, ll);
            ctx_h[idx] = hh;
            ctx_l[idx] = ll;
        }
    }
}

// ---------------------------------------------------------------------------
extern "C" void kernel_launch(void* const* d_in, const int* in_sizes, int n_in,
                              void* d_out, int out_size, void* d_ws, size_t ws_size,
                              hipStream_t stream) {
    const float* x      = (const float*)d_in[0];
    const float* w_attn = (const float*)d_in[1];
    const float* w_proj = (const float*)d_in[2];
    const float* b_proj = (const float*)d_in[3];
    float* out = (float*)d_out;

    // Workspace: bf16 hi/lo planes. ctx aliases x planes (x dead after QKV GEMM).
    __bf16* w = (__bf16*)d_ws;
    __bf16* xh  = w; w += PLANE_X;
    __bf16* xl  = w; w += PLANE_X;
    __bf16* wAh = w; w += PLANE_WA;
    __bf16* wAl = w; w += PLANE_WA;
    __bf16* wPh = w; w += PLANE_WP;
    __bf16* wPl = w; w += PLANE_WP;
    __bf16* qh  = w; w += PLANE_HED;
    __bf16* ql  = w; w += PLANE_HED;
    __bf16* kh  = w; w += PLANE_HED;
    __bf16* kl  = w; w += PLANE_HED;
    __bf16* vh  = w; w += PLANE_HED;
    __bf16* vl  = w; w += PLANE_HED;
    __bf16* ctxh = xh;
    __bf16* ctxl = xl;

    // 0) split inputs into bf16 hi/lo planes (weights transposed to [N][K]).
    split_plain<<<(int)(PLANE_X / 4 / 256), 256, 0, stream>>>(x, xh, xl, (int)(PLANE_X / 4));
    split_transpose<<<(int)(PLANE_WA / 256), 256, 0, stream>>>(w_attn, wAh, wAl, N3);
    split_transpose<<<(int)(PLANE_WP / 256), 256, 0, stream>>>(w_proj, wPh, wPl, CC);

    // 1) qkv = x @ w_attn -> q/k [B,H,T,HD], v [B,H,HD,T] split planes.
    gemm_bf<<<dim3(MM / 64, N3 / 64), 128, 0, stream>>>(
        xh, xl, wAh, wAl, 0, qh, ql, kh, kl, vh, vl, nullptr, nullptr);

    // 2) flash attention -> ctx split planes [B*T, C].
    attn_bf<<<dim3(BQ * NH, TT / 64), 128, 0, stream>>>(
        qh, ql, kh, kl, vh, vl, ctxh, ctxl);

    // 3) out = ctx @ w_proj + b_proj (f32).
    gemm_bf<<<dim3(MM / 64, CC / 64), 128, 0, stream>>>(
        ctxh, ctxl, wPh, wPl, 1, nullptr, nullptr, nullptr, nullptr, nullptr, nullptr,
        b_proj, out);
}
